// GNNModelObj_20366734917650
// MI455X (gfx1250) — compile-verified, matrix operand order checked by reading, not run on hardware
//
#include <hip/hip_runtime.h>
#include <hip/hip_bf16.h>
#include <math.h>

// ---------------------------------------------------------------------------
// GCN (2x GCNConv + mean-pool + linear) for MI455X / gfx1250, wave32 + WMMA.
//   conv1 collapses to a scalar edge aggregation (x is [N,1]).
//   conv2's dense part h1@W2 runs on v_wmma_f32_16x16x32_f16.
//   Edge aggregation is L2-resident gather + f32 atomics (hlin = 51MB < 192MB L2).
// ---------------------------------------------------------------------------

typedef __attribute__((ext_vector_type(16))) _Float16 v16h;
typedef __attribute__((ext_vector_type(8)))  _Float16 v8h;
typedef __attribute__((ext_vector_type(8)))  float    v8f;

#define HIDDEN   128
#define N_GRAPHS 64

// ---- init: deg = 1 (self-loop weight), s = 0, pool sums/cnts = 0 ----------
__global__ void k_init(float* __restrict__ deg, float* __restrict__ s,
                       float* __restrict__ sums, float* __restrict__ cnts, int n) {
  int i = blockIdx.x * blockDim.x + threadIdx.x;
  if (i < n) { deg[i] = 1.0f; s[i] = 0.0f; }
  if (i < N_GRAPHS * HIDDEN) sums[i] = 0.0f;
  if (i < N_GRAPHS) cnts[i] = 0.0f;
}

// ---- deg[dst] += ew -------------------------------------------------------
__global__ void k_deg(const int* __restrict__ ei, const float* __restrict__ ew,
                      float* __restrict__ deg, int E) {
  int e = blockIdx.x * blockDim.x + threadIdx.x;
  if (e < E) atomicAdd(&deg[ei[E + e]], ew[e]);
}

// ---- dinv = rsqrt(deg) ----------------------------------------------------
__global__ void k_dinv(const float* __restrict__ deg, float* __restrict__ dinv, int n) {
  int i = blockIdx.x * blockDim.x + threadIdx.x;
  if (i < n) { float d = deg[i]; dinv[i] = (d > 0.0f) ? rsqrtf(d) : 0.0f; }
}

// ---- layer-1 scalar aggregation: s[dst] += dinv[src]*ew*dinv[dst]*x[src] --
__global__ void k_s(const int* __restrict__ ei, const float* __restrict__ ew,
                    const float* __restrict__ x, const float* __restrict__ dinv,
                    float* __restrict__ s, int E) {
  int e = blockIdx.x * blockDim.x + threadIdx.x;
  if (e < E) {
    int sr = ei[e], ds = ei[E + e];
    atomicAdd(&s[ds], dinv[sr] * ew[e] * dinv[ds] * x[sr]);
  }
}

// ---- h1[n,:] = relu((s[n] + dinv^2 * x[n]) * W1 + b1), stored f16 ---------
__global__ void k_h1(const float* __restrict__ s, const float* __restrict__ x,
                     const float* __restrict__ dinv, const float* __restrict__ W1,
                     const float* __restrict__ b1, _Float16* __restrict__ h1,
                     int nnodes) {
  int n = blockIdx.x;        // padded grid
  int j = threadIdx.x;       // 0..127
  float v = 0.0f;
  if (n < nnodes) {
    float di = dinv[n];
    float sn = s[n] + di * di * x[n];
    v = fmaxf(sn * W1[j] + b1[j], 0.0f);
  }
  h1[(size_t)n * HIDDEN + j] = (_Float16)v;
}

// ---- W2^T in f16 (B operand, column-major-of-B == row-major W2T) ----------
__global__ void k_w2t(const float* __restrict__ W2, _Float16* __restrict__ w2t) {
  int k = blockIdx.x;    // 0..127  (input feature)
  int nn = threadIdx.x;  // 0..127  (output feature)
  w2t[(size_t)nn * HIDDEN + k] = (_Float16)W2[(size_t)k * HIDDEN + nn];
}

// ---- WMMA GEMM: hlin[M,128] = h1[M,128] @ W2[128,128] ---------------------
// One wave -> one 16x16 C tile. A layout (16-bit 16x32): lane m = L&15,
// kbase = (L>>4)*8; halves [0..7] = A[m, k+kbase+j], [8..15] = A[m, k+16+kbase+j].
// B fed from W2T with the mirrored layout (n = L&15, contiguous k runs).
__global__ void k_gemm(const _Float16* __restrict__ h1,
                       const _Float16* __restrict__ w2t,
                       float* __restrict__ hlin) {
  int mtile = blockIdx.x;
  int ntile = blockIdx.y;          // 0..7
  int lane  = threadIdx.x;         // 0..31
  int idx   = lane & 15;
  int kbase = (lane >> 4) * 8;

  const _Float16* arow = h1  + (size_t)(mtile * 16 + idx) * HIDDEN;
  const _Float16* brow = w2t + (size_t)(ntile * 16 + idx) * HIDDEN;

  v8f acc = {};
#pragma unroll
  for (int kk = 0; kk < HIDDEN; kk += 32) {
    v8h alo = *(const v8h*)(arow + kk + kbase);
    v8h ahi = *(const v8h*)(arow + kk + kbase + 16);
    v8h blo = *(const v8h*)(brow + kk + kbase);
    v8h bhi = *(const v8h*)(brow + kk + kbase + 16);
    v16h a, b;
#pragma unroll
    for (int j = 0; j < 8; ++j) {
      a[j] = alo[j]; a[8 + j] = ahi[j];
      b[j] = blo[j]; b[8 + j] = bhi[j];
    }
    acc = __builtin_amdgcn_wmma_f32_16x16x32_f16(
        /*neg_a=*/false, a, /*neg_b=*/false, b,
        /*c_mod=*/(short)0, acc, /*reuse_a=*/false, /*reuse_b=*/false);
  }

  // C/D layout: lanes 0-15 -> M = r, lanes 16-31 -> M = 8 + r; N = lane&15.
  int r0  = (lane >> 4) * 8;
  int col = ntile * 16 + idx;
#pragma unroll
  for (int r = 0; r < 8; ++r) {
    int row = mtile * 16 + r0 + r;
    hlin[(size_t)row * HIDDEN + col] = acc[r];
  }
}

// ---- out2 = dinv^2 * hlin  (self-loop message, also zero-initializes) -----
__global__ void k_out2_init(const float* __restrict__ hlin, const float* __restrict__ dinv,
                            float* __restrict__ out2, int total) {
  int i = blockIdx.x * blockDim.x + threadIdx.x;
  if (i < total) {
    float di = dinv[i >> 7];   // i / 128
    out2[i] = di * di * hlin[i];
  }
}

// ---- edge aggregation: out2[dst,:] += norm_e * hlin[src,:] ----------------
// One wave (32 lanes) per edge, float4 per lane -> global_load_b128 gathers
// from L2-resident hlin, 4x global_atomic_add_f32 scatter.
__global__ void k_agg(const int* __restrict__ ei, const float* __restrict__ ew,
                      const float* __restrict__ dinv, const float* __restrict__ hlin,
                      float* __restrict__ out2, int E) {
  int wave = threadIdx.x >> 5;
  int lane = threadIdx.x & 31;
  int e = blockIdx.x * 8 + wave;
  if (e >= E) return;
  int sr = ei[e], ds = ei[E + e];
  float norm = dinv[sr] * ew[e] * dinv[ds];
  const float4 v = *(const float4*)(hlin + (size_t)sr * HIDDEN + lane * 4);
  float* o = out2 + (size_t)ds * HIDDEN + lane * 4;
  atomicAdd(o + 0, norm * v.x);
  atomicAdd(o + 1, norm * v.y);
  atomicAdd(o + 2, norm * v.z);
  atomicAdd(o + 3, norm * v.w);
}

// ---- finalize (bias + relu) fused with mean-pool accumulation -------------
__global__ void k_pool(const float* __restrict__ out2, const float* __restrict__ b2,
                       const int* __restrict__ batch, float* __restrict__ sums,
                       float* __restrict__ cnts) {
  int n = blockIdx.x;
  int j = threadIdx.x;
  float v = fmaxf(out2[(size_t)n * HIDDEN + j] + b2[j], 0.0f);
  int g = batch[n];
  atomicAdd(&sums[g * HIDDEN + j], v);
  if (j == 0) atomicAdd(&cnts[g], 1.0f);
}

// ---- out[g] = (sums[g,:]/max(cnt,1)) . Wl + bl ----------------------------
__global__ void k_final(const float* __restrict__ sums, const float* __restrict__ cnts,
                        const float* __restrict__ Wl, const float* __restrict__ bl,
                        float* __restrict__ out) {
  int g = blockIdx.x;          // 0..63
  int t = threadIdx.x;         // 0..127
  float c = fmaxf(cnts[g], 1.0f);
  float v = sums[g * HIDDEN + t] / c * Wl[t];
#pragma unroll
  for (int o = 16; o > 0; o >>= 1) v += __shfl_down(v, o, 32);
  __shared__ float red[4];
  if ((t & 31) == 0) red[t >> 5] = v;
  __syncthreads();
  if (t == 0) out[g] = red[0] + red[1] + red[2] + red[3] + bl[0];
}

// ---------------------------------------------------------------------------
extern "C" void kernel_launch(void* const* d_in, const int* in_sizes, int n_in,
                              void* d_out, int out_size, void* d_ws, size_t ws_size,
                              hipStream_t stream) {
  const float* x    = (const float*)d_in[0];
  const int*   ei   = (const int*)d_in[1];     // [2, E]: row0 = src, row1 = dst
  const float* ew   = (const float*)d_in[2];
  const int*   bat  = (const int*)d_in[3];
  const float* W1   = (const float*)d_in[4];
  const float* b1   = (const float*)d_in[5];
  const float* W2   = (const float*)d_in[6];
  const float* b2   = (const float*)d_in[7];
  const float* Wl   = (const float*)d_in[8];
  const float* bl   = (const float*)d_in[9];
  float* out = (float*)d_out;

  const int N = in_sizes[0];         // 100000
  const int E = in_sizes[2];         // 3200000
  const int mtiles = (N + 15) / 16;
  const int Npad = mtiles * 16;

  // Workspace carve-up (256B aligned slices).
  auto align256 = [](size_t v) { return (v + 255) & ~(size_t)255; };
  char* ws = (char*)d_ws;
  size_t off = 0;
  float*    deg  = (float*)(ws + off);  off += align256((size_t)N * 4);
  float*    dinv = (float*)(ws + off);  off += align256((size_t)N * 4);
  float*    s    = (float*)(ws + off);  off += align256((size_t)N * 4);
  _Float16* h1   = (_Float16*)(ws + off); off += align256((size_t)Npad * HIDDEN * 2);
  _Float16* w2t  = (_Float16*)(ws + off); off += align256((size_t)HIDDEN * HIDDEN * 2);
  float*    hlin = (float*)(ws + off);  off += align256((size_t)Npad * HIDDEN * 4);
  float*    out2 = (float*)(ws + off);  off += align256((size_t)N * HIDDEN * 4);
  float*    sums = (float*)(ws + off);  off += align256((size_t)N_GRAPHS * HIDDEN * 4);
  float*    cnts = (float*)(ws + off);  off += align256((size_t)N_GRAPHS * 4);
  (void)ws_size; (void)n_in; (void)out_size;

  const int TB = 256;
  int nb_n = (N + TB - 1) / TB;
  int nb_e = (E + TB - 1) / TB;

  k_init<<<nb_n, TB, 0, stream>>>(deg, s, sums, cnts, N);
  k_deg<<<nb_e, TB, 0, stream>>>(ei, ew, deg, E);
  k_dinv<<<nb_n, TB, 0, stream>>>(deg, dinv, N);
  k_s<<<nb_e, TB, 0, stream>>>(ei, ew, x, dinv, s, E);
  k_h1<<<Npad, HIDDEN, 0, stream>>>(s, x, dinv, W1, b1, h1, N);
  k_w2t<<<HIDDEN, HIDDEN, 0, stream>>>(W2, w2t);
  k_gemm<<<dim3(mtiles, HIDDEN / 16), 32, 0, stream>>>(h1, w2t, hlin);
  {
    int total = N * HIDDEN;
    k_out2_init<<<(total + TB - 1) / TB, TB, 0, stream>>>(hlin, dinv, out2, total);
  }
  k_agg<<<(E + 7) / 8, TB, 0, stream>>>(ei, ew, dinv, hlin, out2, E);
  k_pool<<<N, HIDDEN, 0, stream>>>(out2, b2, bat, sums, cnts);
  k_final<<<N_GRAPHS, HIDDEN, 0, stream>>>(sums, cnts, Wl, bl, out);
}
